// RNNWavefunction_59244778881185
// MI455X (gfx1250) — matrix operand order repaired and checked by reading, not compiled, not run on hardware
//
#include <hip/hip_runtime.h>
#include <stdint.h>

typedef __attribute__((ext_vector_type(16))) __bf16 v16bf;
typedef __attribute__((ext_vector_type(8)))  float  v8f;

#define HD    512
#define G3    1536
#define BT    1024
#define SORB_ 64
#define CHPB  16      // chains per block (WMMA M)

union Frag { uint32_t u[8]; v16bf v; };

__device__ inline unsigned short f2bf(float f) {
  union { float f; uint32_t u; } v; v.f = f;
  uint32_t r = v.u + 0x7FFFu + ((v.u >> 16) & 1u);   // round-to-nearest-even
  return (unsigned short)(r >> 16);
}
__device__ inline float bf2f(unsigned short h) {
  union { float f; uint32_t u; } v; v.u = ((uint32_t)h) << 16;
  return v.f;
}

__device__ inline v8f wmma_bf16(v16bf a, v16bf b, v8f c) {
  // D = A(16x32) x B(32x16) + C, f32 accumulate
  return __builtin_amdgcn_wmma_f32_16x16x32_bf16(false, a, false, b, (short)0, c,
                                                 false, false);
}

// A fragment: row = lane&15 of the [16][512] bf16 LDS matrix, k-chunk kk (32 wide).
// lanes 0-15: K pairs (dwords) {0..7}; lanes 16-31: {8..15}.
__device__ inline v16bf load_a_frag(const uint32_t* row32, int kk, int laneHi) {
  Frag f;
  int base = kk * 16 + (laneHi ? 8 : 0);
#pragma unroll
  for (int j = 0; j < 4; ++j) f.u[j]     = row32[base + j];
#pragma unroll
  for (int j = 0; j < 4; ++j) f.u[4 + j] = row32[base + 4 + j];
  return f.v;
}

// B fragment: B[k][n] = W[n_global][k]; lane holds column n_global = tile*16 + (lane&15).
// lanes 0-15 hold K=kk*32+0..15, lanes 16-31 K=kk*32+16..31 -> 8 contiguous dwords.
__device__ inline v16bf load_b_frag(const uint32_t* __restrict__ wmat32,
                                    int n_global, int kk, int laneHi) {
  const uint32_t* p = wmat32 + (size_t)n_global * (HD / 2) + kk * 16 + (laneHi ? 8 : 0);
  Frag f;
  uint4 lo = *(const uint4*)p;
  uint4 hi = *(const uint4*)(p + 4);
  f.u[0] = lo.x; f.u[1] = lo.y; f.u[2] = lo.z; f.u[3] = lo.w;
  f.u[4] = hi.x; f.u[5] = hi.y; f.u[6] = hi.z; f.u[7] = hi.w;
  return f.v;
}

__device__ inline float sigm(float v) { return 1.f / (1.f + __expf(-v)); }

__global__ void prep_x01(const float* __restrict__ wih0, const float* __restrict__ bih0,
                         float* __restrict__ X0, float* __restrict__ X1) {
  int j = blockIdx.x * 256 + threadIdx.x;
  if (j < G3) {
    X0[j] = wih0[j * 2 + 0] + bih0[j];   // input projection for bit==0 (incl. b_ih0)
    X1[j] = wih0[j * 2 + 1] + bih0[j];   // for bit==1
  }
}

__global__ void cvt_bf16(const float* __restrict__ in, unsigned short* __restrict__ out, int n) {
  int i = blockIdx.x * 256 + threadIdx.x;
  if (i < n) out[i] = f2bf(in[i]);
}

__global__ __launch_bounds__(256, 1)
void gru_chains(const int*   __restrict__ x,
                const float* __restrict__ bhh0,
                const float* __restrict__ bih1,
                const float* __restrict__ bhh1,
                const float* __restrict__ fcw,
                const float* __restrict__ fcb,
                const float* __restrict__ X0,
                const float* __restrict__ X1,
                const uint32_t* __restrict__ whh0,
                const uint32_t* __restrict__ wih1,
                const uint32_t* __restrict__ whh1,
                float* __restrict__ part) {
  __shared__ unsigned short h0s[CHPB][HD];   // bf16 copy of layer-0 state (WMMA A)
  __shared__ unsigned short h1s[CHPB][HD];   // bf16 copy of layer-1 state
  __shared__ int   bits[CHPB];
  __shared__ float ys[CHPB][2];

  const int tid    = threadIdx.x;
  const int wave   = tid >> 5;         // 8 waves; wave owns j in [wave*64, wave*64+64)
  const int lane   = tid & 31;
  const int laneHi = lane >> 4;
  const int ln     = lane & 15;
  const int chain0 = blockIdx.x * CHPB;
  const int jbase  = wave * 64;

  for (int i = tid; i < CHPB * HD; i += 256) {
    ((unsigned short*)h0s)[i] = 0;
    ((unsigned short*)h1s)[i] = 0;
  }
  float h0r[4][8], h1r[4][8];          // f32 master recurrent state, register-resident
#pragma unroll
  for (int q = 0; q < 4; ++q)
#pragma unroll
    for (int m = 0; m < 8; ++m) { h0r[q][m] = 0.f; h1r[q][m] = 0.f; }

  const v8f vzero = {0.f, 0.f, 0.f, 0.f, 0.f, 0.f, 0.f, 0.f};

  for (int t = 0; t < BT; ++t) {
    if (tid < CHPB) bits[tid] = (1 - x[t * SORB_ + chain0 + tid]) >> 1;
    __syncthreads();

    // ---------------- layer 0: gates = h0 @ whh0^T ----------------
    v8f aR[4], aZ[4], aN[4];
#pragma unroll
    for (int q = 0; q < 4; ++q) { aR[q] = vzero; aZ[q] = vzero; aN[q] = vzero; }
    for (int kk = 0; kk < HD / 32; ++kk) {
      v16bf a = load_a_frag((const uint32_t*)&h0s[ln][0], kk, laneHi);
#pragma unroll
      for (int q = 0; q < 4; ++q) {
        int n0 = jbase + q * 16 + ln;
        aR[q] = wmma_bf16(a, load_b_frag(whh0, n0,         kk, laneHi), aR[q]);
        aZ[q] = wmma_bf16(a, load_b_frag(whh0, 512 + n0,   kk, laneHi), aZ[q]);
        aN[q] = wmma_bf16(a, load_b_frag(whh0, 1024 + n0,  kk, laneHi), aN[q]);
      }
    }
    __syncthreads();

    // layer-0 GRU combine (wave-local: lane owns column j, vgpr slot owns chain)
#pragma unroll
    for (int q = 0; q < 4; ++q) {
      int j = jbase + q * 16 + ln;
      float b_r = bhh0[j], b_z = bhh0[512 + j], b_n = bhh0[1024 + j];
      float x0r = X0[j],        x1r = X1[j];
      float x0z = X0[512 + j],  x1z = X1[512 + j];
      float x0n = X0[1024 + j], x1n = X1[1024 + j];
#pragma unroll
      for (int m8 = 0; m8 < 8; ++m8) {
        int m  = m8 + laneHi * 8;
        int bc = bits[m];
        float xr = bc ? x1r : x0r;
        float xz = bc ? x1z : x0z;
        float xn = bc ? x1n : x0n;
        float r = sigm(xr + aR[q][m8] + b_r);
        float z = sigm(xz + aZ[q][m8] + b_z);
        float n = tanhf(xn + r * (aN[q][m8] + b_n));
        float hn = (1.f - z) * n + z * h0r[q][m8];
        h0r[q][m8] = hn;
        h0s[m][j]  = f2bf(hn);
      }
    }
    __syncthreads();

    // ---------------- layer 1: r,z accumulate both GEMMs; xn/hn separate ----------
    v8f cR[4], cZ[4], cXN[4], cHN[4];
#pragma unroll
    for (int q = 0; q < 4; ++q) { cR[q] = vzero; cZ[q] = vzero; cXN[q] = vzero; cHN[q] = vzero; }
    for (int kk = 0; kk < HD / 32; ++kk) {
      v16bf a0 = load_a_frag((const uint32_t*)&h0s[ln][0], kk, laneHi);  // new h0 (input)
      v16bf a1 = load_a_frag((const uint32_t*)&h1s[ln][0], kk, laneHi);  // old h1
#pragma unroll
      for (int q = 0; q < 4; ++q) {
        int n0 = jbase + q * 16 + ln;
        cR[q]  = wmma_bf16(a0, load_b_frag(wih1, n0,        kk, laneHi), cR[q]);
        cR[q]  = wmma_bf16(a1, load_b_frag(whh1, n0,        kk, laneHi), cR[q]);
        cZ[q]  = wmma_bf16(a0, load_b_frag(wih1, 512 + n0,  kk, laneHi), cZ[q]);
        cZ[q]  = wmma_bf16(a1, load_b_frag(whh1, 512 + n0,  kk, laneHi), cZ[q]);
        cXN[q] = wmma_bf16(a0, load_b_frag(wih1, 1024 + n0, kk, laneHi), cXN[q]);
        cHN[q] = wmma_bf16(a1, load_b_frag(whh1, 1024 + n0, kk, laneHi), cHN[q]);
      }
    }
    __syncthreads();

#pragma unroll
    for (int q = 0; q < 4; ++q) {
      int j = jbase + q * 16 + ln;
      float br = bih1[j] + bhh1[j];
      float bz = bih1[512 + j] + bhh1[512 + j];
      float bxn = bih1[1024 + j];
      float bhn = bhh1[1024 + j];
#pragma unroll
      for (int m8 = 0; m8 < 8; ++m8) {
        int m = m8 + laneHi * 8;
        float r = sigm(cR[q][m8] + br);
        float z = sigm(cZ[q][m8] + bz);
        float n = tanhf(cXN[q][m8] + bxn + r * (cHN[q][m8] + bhn));
        float hn = (1.f - z) * n + z * h1r[q][m8];
        h1r[q][m8] = hn;
        h1s[m][j]  = f2bf(hn);
      }
    }
    __syncthreads();

    // ---------------- FC(2x512) + log-softmax contribution -------------------
    {
      int pair = tid >> 3, sub = tid & 7;      // 32 (chain,label) pairs x 8 lanes
      int ch = pair >> 1, lbl = pair & 1;
      float p = 0.f;
      for (int k = sub; k < HD; k += 8)
        p += bf2f(h1s[ch][k]) * fcw[lbl * HD + k];
      p += __shfl_down(p, 4, 8);
      p += __shfl_down(p, 2, 8);
      p += __shfl_down(p, 1, 8);
      if (sub == 0) ys[ch][lbl] = p;
    }
    __syncthreads();
    if (tid < CHPB) {
      float y0 = ys[tid][0] + fcb[0];
      float y1 = ys[tid][1] + fcb[1];
      float mx = fmaxf(y0, y1);
      float lse = mx + logf(__expf(y0 - mx) + __expf(y1 - mx));
      float sel = bits[tid] ? y1 : y0;
      part[(size_t)(chain0 + tid) * BT + t] = 0.5f * (sel - lse);
    }
    __syncthreads();
  }
}

__global__ void finalize(const float* __restrict__ part, float* __restrict__ out) {
  int t = blockIdx.x * 256 + threadIdx.x;
  if (t < BT) {
    float s = 0.f;
    for (int i = 0; i < SORB_; ++i) s += part[(size_t)i * BT + t];
    out[t] = __expf(s);
  }
}

extern "C" void kernel_launch(void* const* d_in, const int* in_sizes, int n_in,
                              void* d_out, int out_size, void* d_ws, size_t ws_size,
                              hipStream_t stream) {
  const int*   x    = (const int*)  d_in[0];
  const float* wih0 = (const float*)d_in[1];
  const float* whh0 = (const float*)d_in[2];
  const float* bih0 = (const float*)d_in[3];
  const float* bhh0 = (const float*)d_in[4];
  const float* wih1 = (const float*)d_in[5];
  const float* whh1 = (const float*)d_in[6];
  const float* bih1 = (const float*)d_in[7];
  const float* bhh1 = (const float*)d_in[8];
  const float* fcw  = (const float*)d_in[9];
  const float* fcb  = (const float*)d_in[10];

  char* ws = (char*)d_ws;
  float*          X0    = (float*)ws;          ws += (size_t)G3 * 4;        // 6 KB
  float*          X1    = (float*)ws;          ws += (size_t)G3 * 4;
  unsigned short* whh0b = (unsigned short*)ws; ws += (size_t)G3 * HD * 2;   // 1.5 MB
  unsigned short* wih1b = (unsigned short*)ws; ws += (size_t)G3 * HD * 2;
  unsigned short* whh1b = (unsigned short*)ws; ws += (size_t)G3 * HD * 2;
  float*          part  = (float*)ws;          ws += (size_t)SORB_ * BT * 4;

  prep_x01<<<(G3 + 255) / 256, 256, 0, stream>>>(wih0, bih0, X0, X1);
  const int NW = G3 * HD;
  cvt_bf16<<<(NW + 255) / 256, 256, 0, stream>>>(whh0, whh0b, NW);
  cvt_bf16<<<(NW + 255) / 256, 256, 0, stream>>>(wih1, wih1b, NW);
  cvt_bf16<<<(NW + 255) / 256, 256, 0, stream>>>(whh1, whh1b, NW);

  gru_chains<<<SORB_ / CHPB, 256, 0, stream>>>(
      x, bhh0, bih1, bhh1, fcw, fcb, X0, X1,
      (const uint32_t*)whh0b, (const uint32_t*)wih1b, (const uint32_t*)whh1b, part);

  finalize<<<(BT + 255) / 256, 256, 0, stream>>>(part, (float*)d_out);
}
